// ResMambaBlock_15539191677176
// MI455X (gfx1250) — matrix operand order repaired and checked by reading, not compiled, not run on hardware
//
#include <hip/hip_runtime.h>

#define Bq 2
#define Cq 96
#define Hq 64
#define Wq 64
#define Lq 4096
#define Dq 192
#define Nq 16
#define Rq 6
#define Kq 4
#define CDq 38           // R + 2N
#define TLq 64
#define EPSq 1e-5f

typedef __attribute__((ext_vector_type(16))) _Float16 v16h;
typedef __attribute__((ext_vector_type(8)))  float    v8f;

#if defined(__has_builtin)
#  if __has_builtin(__builtin_amdgcn_tensor_load_to_lds) && \
      __has_builtin(__builtin_amdgcn_s_wait_tensorcnt)
#    define USE_TDM 1
#  endif
#endif
#ifndef USE_TDM
#  define USE_TDM 0
#endif

__device__ __forceinline__ float silu_f(float x) { return x / (1.f + __expf(-x)); }
__device__ __forceinline__ float softplus_f(float x) {
  return (x > 20.f) ? x : log1pf(__expf(x));
}

#if USE_TDM
typedef __attribute__((ext_vector_type(4))) unsigned int u32x4;
typedef __attribute__((ext_vector_type(4))) int          i32x4;
typedef __attribute__((ext_vector_type(8))) int          i32x8;

// Issue one TDM 2-D tile load: CDq rows x TLq floats, global row stride Lq floats,
// packed contiguously into LDS at lds_byte_addr. D# layout per CDNA5 ISA ch.8.
__device__ __forceinline__ void tdm_load_tile(unsigned lds_byte_addr,
                                              const float* gsrc) {
  unsigned long long ga = (unsigned long long)(size_t)gsrc;
  u32x4 g0;
  g0.x = 1u;                                            // count=1, user descriptor
  g0.y = lds_byte_addr;                                 // lds_addr
  g0.z = (unsigned)(ga & 0xFFFFFFFFu);                  // global_addr[31:0]
  g0.w = (unsigned)((ga >> 32) & 0x01FFFFFFu) | (2u << 30);  // addr[56:32], type=2
  i32x8 g1;
  g1[0] = 0x00020000;                                   // wg_mask=0, data_size=4B
  g1[1] = (int)((Lq & 0xFFFF) << 16);                   // tensor_dim0[15:0]
  g1[2] = (int)(((unsigned)Lq >> 16) | ((unsigned)CDq << 16)); // dim0 hi | dim1 lo
  g1[3] = (int)((unsigned)TLq << 16);                   // dim1 hi | tile_dim0=64
  g1[4] = CDq;                                          // tile_dim1=38, tile_dim2=0
  g1[5] = Lq;                                           // tensor_dim0_stride lo
  g1[6] = 0;                                            // stride hi | dim1_stride lo
  g1[7] = 0;
  i32x4 gz = {0, 0, 0, 0};
#if __clang_major__ >= 23
  i32x8 gz8 = {0, 0, 0, 0, 0, 0, 0, 0};
  __builtin_amdgcn_tensor_load_to_lds(g0, g1, gz, gz, gz8, 0);
#else
  __builtin_amdgcn_tensor_load_to_lds(g0, g1, gz, gz, 0);
#endif
}
#endif  // USE_TDM

// ---------------- K1: depthwise 3x3 conv on input (BCHW -> BHWC) ----------------
__global__ void k_dwconv1(const float* __restrict__ x, const float* __restrict__ w,
                          const float* __restrict__ bias, float* __restrict__ hb) {
  int idx = blockIdx.x * blockDim.x + threadIdx.x;
  if (idx >= Bq * Hq * Wq * Cq) return;
  int c = idx % Cq;
  int p = idx / Cq;
  int wcol = p % Wq; int t = p / Wq; int hrow = t % Hq; int b = t / Hq;
  const float* xp = x + ((size_t)b * Cq + c) * (Hq * Wq);
  const float* wp = w + c * 9;
  float acc = bias[c];
#pragma unroll
  for (int kh = 0; kh < 3; ++kh) {
    int hh = hrow + kh - 1;
    if ((unsigned)hh >= (unsigned)Hq) continue;
#pragma unroll
    for (int kw = 0; kw < 3; ++kw) {
      int ww = wcol + kw - 1;
      if ((unsigned)ww >= (unsigned)Wq) continue;
      acc += wp[kh * 3 + kw] * xp[hh * Wq + ww];
    }
  }
  hb[idx] = acc;  // BHWC
}

// ---------------- K2: LayerNorm over C, emit f16 for WMMA ----------------
__global__ void k_ln_f16(const float* __restrict__ hb, const float* __restrict__ lw,
                         const float* __restrict__ lb, _Float16* __restrict__ lnh) {
  __shared__ float s1[128], s2[128];
  int pix = blockIdx.x;
  int tid = threadIdx.x;  // 128 threads, 96 active
  float v = (tid < Cq) ? hb[(size_t)pix * Cq + tid] : 0.f;
  s1[tid] = v; s2[tid] = v * v;
  __syncthreads();
  for (int st = 64; st > 0; st >>= 1) {
    if (tid < st) { s1[tid] += s1[tid + st]; s2[tid] += s2[tid + st]; }
    __syncthreads();
  }
  float m = s1[0] / (float)Cq;
  float var = s2[0] / (float)Cq - m * m;
  float rinv = rsqrtf(var + EPSq);
  if (tid < Cq)
    lnh[(size_t)pix * Cq + tid] = (_Float16)((v - m) * rinv * lw[tid] + lb[tid]);
}

// ---------------- f32 -> f16 weight cast ----------------
__global__ void k_cvt_f16(const float* __restrict__ a, _Float16* __restrict__ o, int n) {
  int i = blockIdx.x * blockDim.x + threadIdx.x;
  if (i < n) o[i] = (_Float16)a[i];
}

// ---------------- WMMA GEMM: C[M,N] = A[M,K] * W[N,K]^T (f16 in, f32 out) --------
template <int MM, int NN, int KK>
__global__ void k_wmma_gemm(const _Float16* __restrict__ A,
                            const _Float16* __restrict__ Wt,
                            float* __restrict__ Cc) {
  int wave = blockIdx.x * (blockDim.x >> 5) + (threadIdx.x >> 5);
  int lane = threadIdx.x & 31;
  const int TN = NN / 16;
  int tM = wave / TN, tN = wave % TN;
  if (tM >= MM / 16) return;
  int r = lane & 15;      // row (A) / col (B,C)
  int hs = lane >> 4;     // half-wave select
  v8f acc = {};
  for (int kk = 0; kk < KK; kk += 32) {
    v16h a, bm;
#pragma unroll
    for (int j = 0; j < 8; ++j) {
      int ka = ((j < 4) ? 2 * j : 2 * j + 8) + hs * 8;
      const _Float16* ap = A + (size_t)(tM * 16 + r) * KK + kk + ka;
      a[2 * j] = ap[0]; a[2 * j + 1] = ap[1];
      int kb = ((j < 4) ? 2 * j : 2 * j + 8) + hs * 8;
      const _Float16* bp = Wt + (size_t)(tN * 16 + r) * KK + kk + kb;
      bm[2 * j] = bp[0]; bm[2 * j + 1] = bp[1];
    }
    acc = __builtin_amdgcn_wmma_f32_16x16x32_f16(false, a, false, bm,
                                                 (short)0, acc, false, false);
  }
#pragma unroll
  for (int i = 0; i < 8; ++i) {
    int m = tM * 16 + i + hs * 8;
    Cc[(size_t)m * NN + tN * 16 + r] = acc[i];
  }
}

// ---------------- K4: depthwise 3x3 conv on xi (from xz BHW[2D]) + SiLU ----------
__global__ void k_conv2_silu(const float* __restrict__ xz, const float* __restrict__ w,
                             const float* __restrict__ bias, float* __restrict__ xs) {
  int idx = blockIdx.x * blockDim.x + threadIdx.x;  // over B*D*L
  if (idx >= Bq * Dq * Lq) return;
  int l = idx % Lq; int t = idx / Lq; int d = t % Dq; int b = t / Dq;
  int wcol = l % Wq, hrow = l / Wq;
  const float* wp = w + d * 9;
  float acc = bias[d];
#pragma unroll
  for (int kh = 0; kh < 3; ++kh) {
    int hh = hrow + kh - 1;
    if ((unsigned)hh >= (unsigned)Hq) continue;
#pragma unroll
    for (int kw = 0; kw < 3; ++kw) {
      int ww = wcol + kw - 1;
      if ((unsigned)ww >= (unsigned)Wq) continue;
      acc += wp[kh * 3 + kw] *
             xz[((size_t)b * Lq + hh * Wq + ww) * (2 * Dq) + d];
    }
  }
  acc = silu_f(acc);
  xs[((size_t)(b * Kq + 0) * Dq + d) * Lq + l] = acc;
}

// ---------------- K5: build scan directions k=1,2,3 from k=0 ----------------
__global__ void k_scan_dirs(float* __restrict__ xs) {
  int idx = blockIdx.x * blockDim.x + threadIdx.x;  // over B*D*L
  if (idx >= Bq * Dq * Lq) return;
  int l = idx % Lq; int t = idx / Lq; int d = t % Dq; int b = t / Dq;
  float v = xs[((size_t)(b * Kq + 0) * Dq + d) * Lq + l];
  int hrow = l / Wq, wcol = l % Wq;
  int l2 = wcol * Hq + hrow;
  xs[((size_t)(b * Kq + 1) * Dq + d) * Lq + l2] = v;
  xs[((size_t)(b * Kq + 2) * Dq + d) * Lq + (Lq - 1 - l)] = v;
  xs[((size_t)(b * Kq + 3) * Dq + d) * Lq + (Lq - 1 - l2)] = v;
}

// ---------------- K6: x_proj  x_dbl[b,k,c,l] = sum_d W[k,c,d]*xs[b,k,d,l] --------
__global__ void k_xproj(const float* __restrict__ xs, const float* __restrict__ xw,
                        float* __restrict__ xdbl) {
  __shared__ float lw[CDq * Dq];
  int bk = blockIdx.x / (Lq / 256);
  int lch = blockIdx.x % (Lq / 256);
  int b = bk / Kq, k = bk % Kq;
  for (int i = threadIdx.x; i < CDq * Dq; i += 256)
    lw[i] = xw[(size_t)k * CDq * Dq + i];
  __syncthreads();
  int l = lch * 256 + threadIdx.x;
  float acc[CDq];
#pragma unroll
  for (int c = 0; c < CDq; ++c) acc[c] = 0.f;
  const float* xp = xs + ((size_t)(b * Kq + k) * Dq) * Lq + l;
  for (int d = 0; d < Dq; ++d) {
    float v = xp[(size_t)d * Lq];
#pragma unroll
    for (int c = 0; c < CDq; ++c) acc[c] += lw[c * Dq + d] * v;
  }
  float* op = xdbl + ((size_t)(b * Kq + k) * CDq) * Lq + l;
#pragma unroll
  for (int c = 0; c < CDq; ++c) op[(size_t)c * Lq] = acc[c];
}

// ---------------- K7: fused dt_proj + softplus + selective scan ----------------
// one block per (b,k); one lane per channel d; N=16 states in registers.
// B/C/dt tiles staged to LDS; TDM double-buffered when available.
__global__ void k_scan(const float* __restrict__ xs, const float* __restrict__ xdbl,
                       const float* __restrict__ dtw_g, const float* __restrict__ dtb_g,
                       const float* __restrict__ Alog, const float* __restrict__ Dsg,
                       float* __restrict__ outy) {
  __shared__ float sd[2][CDq][TLq];
  int bk = blockIdx.x; int b = bk / Kq, k = bk % Kq;
  int d = threadIdx.x;  // 192
  float dtw[Rq], Arow[Nq], hst[Nq];
#pragma unroll
  for (int r = 0; r < Rq; ++r) dtw[r] = dtw_g[((size_t)k * Dq + d) * Rq + r];
  float dtb = dtb_g[k * Dq + d];
  float Dval = Dsg[k * Dq + d];
#pragma unroll
  for (int n = 0; n < Nq; ++n) {
    Arow[n] = -__expf(Alog[((size_t)k * Dq + d) * Nq + n]);
    hst[n] = 0.f;
  }
  const float* up = xs + ((size_t)(b * Kq + k) * Dq + d) * Lq;
  float* yp = outy + ((size_t)(b * Kq + k) * Dq + d) * Lq;
  const float* xdp = xdbl + ((size_t)(b * Kq + k) * CDq) * Lq;
  const int NCH = Lq / TLq;

#if USE_TDM
  bool issuer = (threadIdx.x >> 5) == 0;  // wave 0 owns the TDM pipeline
  unsigned lds0 =
      (unsigned)(unsigned long long)(__attribute__((address_space(3))) char*)
          (void*)&sd[0][0][0];
  const unsigned bufBytes = (unsigned)(CDq * TLq * sizeof(float));
  if (issuer) tdm_load_tile(lds0, xdp);                 // prime chunk 0 -> buf 0
  for (int ci = 0; ci < NCH; ++ci) {
    int l0 = ci * TLq;
    int buf = ci & 1;
    if (issuer) __builtin_amdgcn_s_wait_tensorcnt(0);   // chunk ci landed in LDS
    __syncthreads();                                    // publish to all waves
    if (issuer && (ci + 1 < NCH))                       // prefetch chunk ci+1
      tdm_load_tile(lds0 + (unsigned)(buf ^ 1) * bufBytes, xdp + (l0 + TLq));
    if (l0 + TLq < Lq) __builtin_prefetch(up + l0 + TLq, 0, 1);
    const float (*sc)[TLq] = sd[buf];
    for (int j = 0; j < TLq; ++j) {
      float u = up[l0 + j];
      float dt = dtb;
#pragma unroll
      for (int r = 0; r < Rq; ++r) dt += dtw[r] * sc[r][j];
      float delta = softplus_f(dt);
      float du = delta * u;
      float y = 0.f;
#pragma unroll
      for (int n = 0; n < Nq; ++n) {
        float dA = __expf(delta * Arow[n]);
        hst[n] = hst[n] * dA + du * sc[Rq + n][j];
        y += hst[n] * sc[Rq + Nq + n][j];
      }
      yp[l0 + j] = y + Dval * u;
    }
  }
#else
  for (int l0 = 0; l0 < Lq; l0 += TLq) {
    __syncthreads();
    for (int i = threadIdx.x; i < CDq * TLq; i += blockDim.x) {
      int c = i / TLq, j = i % TLq;
      sd[0][c][j] = xdp[(size_t)c * Lq + l0 + j];
    }
    __syncthreads();
    if (l0 + TLq < Lq) __builtin_prefetch(up + l0 + TLq, 0, 1);
    for (int j = 0; j < TLq; ++j) {
      float u = up[l0 + j];
      float dt = dtb;
#pragma unroll
      for (int r = 0; r < Rq; ++r) dt += dtw[r] * sd[0][r][j];
      float delta = softplus_f(dt);
      float du = delta * u;
      float y = 0.f;
#pragma unroll
      for (int n = 0; n < Nq; ++n) {
        float dA = __expf(delta * Arow[n]);
        hst[n] = hst[n] * dA + du * sd[0][Rq + n][j];
        y += hst[n] * sd[0][Rq + Nq + n][j];
      }
      yp[l0 + j] = y + Dval * u;
    }
  }
#endif
}

// ---------------- K8: merge 4 directions + out_norm LN + SiLU gate -> f16 --------
__global__ void k_combine(const float* __restrict__ outy, const float* __restrict__ xz,
                          const float* __restrict__ onw, const float* __restrict__ onb,
                          _Float16* __restrict__ yg) {
  __shared__ float s1[256], s2[256];
  int pix = blockIdx.x;  // b*L + l
  int b = pix / Lq, l = pix % Lq;
  int tid = threadIdx.x;  // 256 threads, 192 active
  int hrow = l / Wq, wcol = l % Wq;
  int lT = wcol * Hq + hrow;
  float y = 0.f;
  if (tid < Dq) {
    size_t s = ((size_t)(b * Kq) * Dq + tid) * Lq;
    size_t stride = (size_t)Dq * Lq;
    y = outy[s + l] + outy[s + stride + lT] +
        outy[s + 2 * stride + (Lq - 1 - l)] +
        outy[s + 3 * stride + (Lq - 1 - lT)];
  }
  s1[tid] = y; s2[tid] = y * y;
  __syncthreads();
  for (int st = 128; st > 0; st >>= 1) {
    if (tid < st) { s1[tid] += s1[tid + st]; s2[tid] += s2[tid + st]; }
    __syncthreads();
  }
  float m = s1[0] / (float)Dq;
  float var = s2[0] / (float)Dq - m * m;
  float rinv = rsqrtf(var + EPSq);
  if (tid < Dq) {
    float yn = (y - m) * rinv * onw[tid] + onb[tid];
    float z = xz[(size_t)pix * (2 * Dq) + Dq + tid];
    yg[(size_t)pix * Dq + tid] = (_Float16)(yn * silu_f(z));
  }
}

// ---------------- K11: residual + instance-norm + LeakyReLU + scaled skip --------
__global__ void k_final(const float* __restrict__ hb, const float* __restrict__ ss,
                        const float* __restrict__ x, const float* __restrict__ inw,
                        const float* __restrict__ inb, const float* __restrict__ scale,
                        float* __restrict__ out) {
  __shared__ float s1[256], s2[256];
  int bc = blockIdx.x; int b = bc / Cq, c = bc % Cq;
  int tid = threadIdx.x;
  float sum = 0.f, sq = 0.f;
  for (int l = tid; l < Lq; l += 256) {
    size_t i = ((size_t)b * Lq + l) * Cq + c;
    float v = hb[i] + ss[i];
    sum += v; sq += v * v;
  }
  s1[tid] = sum; s2[tid] = sq;
  __syncthreads();
  for (int st = 128; st > 0; st >>= 1) {
    if (tid < st) { s1[tid] += s1[tid + st]; s2[tid] += s2[tid + st]; }
    __syncthreads();
  }
  float m = s1[0] / (float)Lq;
  float var = s2[0] / (float)Lq - m * m;
  float rinv = rsqrtf(var + EPSq);
  float sc = scale[0];
  for (int l = tid; l < Lq; l += 256) {
    size_t i = ((size_t)b * Lq + l) * Cq + c;
    float v = hb[i] + ss[i];
    v = (v - m) * rinv * inw[c] + inb[c];
    v = (v >= 0.f) ? v : 0.01f * v;
    out[((size_t)b * Cq + c) * Lq + l] = v + x[((size_t)b * Cq + c) * Lq + l] * sc;
  }
}

extern "C" void kernel_launch(void* const* d_in, const int* in_sizes, int n_in,
                              void* d_out, int out_size, void* d_ws, size_t ws_size,
                              hipStream_t stream) {
  (void)in_sizes; (void)n_in; (void)out_size; (void)ws_size;
  const float* x          = (const float*)d_in[0];
  const float* conv_w     = (const float*)d_in[1];
  const float* conv_b     = (const float*)d_in[2];
  const float* inorm_w    = (const float*)d_in[3];
  const float* inorm_b    = (const float*)d_in[4];
  const float* scale      = (const float*)d_in[5];
  const float* ln_w       = (const float*)d_in[6];
  const float* ln_b       = (const float*)d_in[7];
  const float* in_proj_w  = (const float*)d_in[8];
  const float* conv2_w    = (const float*)d_in[9];
  const float* conv2_b    = (const float*)d_in[10];
  const float* x_proj_w   = (const float*)d_in[11];
  const float* dt_proj_w  = (const float*)d_in[12];
  const float* dt_proj_b  = (const float*)d_in[13];
  const float* A_logs     = (const float*)d_in[14];
  const float* Ds         = (const float*)d_in[15];
  const float* out_norm_w = (const float*)d_in[16];
  const float* out_norm_b = (const float*)d_in[17];
  const float* out_proj_w = (const float*)d_in[18];
  float* out = (float*)d_out;

  char* base = (char*)d_ws;
  size_t off = 0;
  auto carve = [&](size_t bytes) -> void* {
    void* q = base + off;
    off = (off + bytes + 255) & ~(size_t)255;
    return q;
  };
  float*    hb    = (float*)carve((size_t)Bq * Lq * Cq * 4);
  _Float16* lnh   = (_Float16*)carve((size_t)Bq * Lq * Cq * 2);
  _Float16* w_in  = (_Float16*)carve((size_t)2 * Dq * Cq * 2);
  _Float16* w_out = (_Float16*)carve((size_t)Cq * Dq * 2);
  float*    xz    = (float*)carve((size_t)Bq * Lq * 2 * Dq * 4);
  float*    xs    = (float*)carve((size_t)Bq * Kq * Dq * Lq * 4);
  float*    xdbl  = (float*)carve((size_t)Bq * Kq * CDq * Lq * 4);
  float*    outy  = (float*)carve((size_t)Bq * Kq * Dq * Lq * 4);
  _Float16* yg    = (_Float16*)carve((size_t)Bq * Lq * Dq * 2);
  float*    ssb   = (float*)carve((size_t)Bq * Lq * Cq * 4);

  k_dwconv1<<<(Bq * Lq * Cq) / 256, 256, 0, stream>>>(x, conv_w, conv_b, hb);
  k_ln_f16<<<Bq * Lq, 128, 0, stream>>>(hb, ln_w, ln_b, lnh);
  k_cvt_f16<<<(2 * Dq * Cq + 255) / 256, 256, 0, stream>>>(in_proj_w, w_in, 2 * Dq * Cq);
  k_cvt_f16<<<(Cq * Dq + 255) / 256, 256, 0, stream>>>(out_proj_w, w_out, Cq * Dq);
  {  // in_proj: M=8192, N=384, K=96 -> 12288 waves
    int waves = (Bq * Lq / 16) * (2 * Dq / 16);
    k_wmma_gemm<Bq * Lq, 2 * Dq, Cq><<<waves / 8, 256, 0, stream>>>(lnh, w_in, xz);
  }
  k_conv2_silu<<<(Bq * Dq * Lq) / 256, 256, 0, stream>>>(xz, conv2_w, conv2_b, xs);
  k_scan_dirs<<<(Bq * Dq * Lq) / 256, 256, 0, stream>>>(xs);
  k_xproj<<<Bq * Kq * (Lq / 256), 256, 0, stream>>>(xs, x_proj_w, xdbl);
  k_scan<<<Bq * Kq, Dq, 0, stream>>>(xs, xdbl, dt_proj_w, dt_proj_b, A_logs, Ds, outy);
  k_combine<<<Bq * Lq, 256, 0, stream>>>(outy, xz, out_norm_w, out_norm_b, yg);
  {  // out_proj: M=8192, N=96, K=192 -> 3072 waves
    int waves = (Bq * Lq / 16) * (Cq / 16);
    k_wmma_gemm<Bq * Lq, Cq, Dq><<<waves / 8, 256, 0, stream>>>(yg, w_out, ssb);
  }
  k_final<<<Bq * Cq, 256, 0, stream>>>(hb, ssb, x, inorm_w, inorm_b, scale, out);
}
